// Dynamic_Q_ResMLP24_89859305767555
// MI455X (gfx1250) — compile-verified, matrix operand order checked by reading, not run on hardware
//
#include <hip/hip_runtime.h>
#include <math.h>

// ---------------- model dims ----------------
#define DIM     384
#define PATCHES 196
#define HID     1536
#define NCLS    1000
#define NBLK    24
#define BATCH   64
#define ROWS    (BATCH * PATCHES)   // 12544
#define CONVK   768                 // 3*16*16

typedef _Float16 h8  __attribute__((ext_vector_type(8)));
typedef _Float16 h16 __attribute__((ext_vector_type(16)));
typedef float    f8  __attribute__((ext_vector_type(8)));
typedef unsigned int uint32x4 __attribute__((ext_vector_type(4)));
typedef int          int32x4  __attribute__((ext_vector_type(4)));
typedef int          int32x8  __attribute__((ext_vector_type(8)));

// ---------------- GEMM config ----------------
#define BM 64
#define BN 64
#define BK 32
#define LDS_STRIDE 40                    // halves: 64B data + 16B pad -> conflict-free frags
#define TILE_HALVES (BM * LDS_STRIDE)    // 2560 halves = 5120 B per tile
#define STAGE_BYTES (2 * TILE_HALVES * 2)  // A+B per stage = 10240 B
#define GEMM_LDS_BYTES (2 * STAGE_BYTES)   // double buffered = 20480 B

enum { EP_BIAS_F32 = 0, EP_CROSS = 1, EP_GELU_F16 = 2, EP_GAMMA_RES = 3 };

__device__ inline h16 cat16(h8 lo, h8 hi) {
  h16 r;
#pragma unroll
  for (int i = 0; i < 8; ++i) { r[i] = lo[i]; r[i + 8] = hi[i]; }
  return r;
}

// ---- Tensor Data Mover: stream a BMxBK f16 tile (row stride = rowStride elems)
// into LDS at byte offset ldsByteOff, padding each 64B row by 16B (-> 40-half
// LDS stride), zero-filling reads past remRows/remCols (ISA 8.2: OOB reads = 0).
__device__ inline void tdm_load_tile(const _Float16* gbase, int rowStride,
                                     int remRows, int remCols,
                                     unsigned ldsByteOff) {
  const unsigned long long ga = (unsigned long long)(const void*)gbase;
  uint32x4 g0;
  g0[0] = 1u;                                       // count=1 (valid user D#)
  g0[1] = ldsByteOff;                               // lds_addr [63:32]
  g0[2] = (unsigned)(ga & 0xFFFFFFFFu);             // global_addr lo
  g0[3] = (unsigned)((ga >> 32) & 0x01FFFFFFu)      // global_addr[56:32]
          | 0x80000000u;                            // type=2 ("image") [127:126]
  const unsigned tdim0 = (unsigned)remCols;         // valid cols from tile start
  const unsigned tdim1 = (unsigned)remRows;         // valid rows from tile start
  const unsigned long long s0 = (unsigned long long)(unsigned)rowStride;
  int32x8 g1;
  g1[0] = (int)((1u << 16)        // data_size=1 -> 2 bytes
              | (1u << 20)        // pad_enable
              | (3u << 22)        // pad_interval=3 -> every 16 DWORDs (one 64B row)
              | (3u << 25));      // pad_amount=3 -> 4 DWORDs (16B) pad
  g1[1] = (int)((tdim0 & 0xFFFFu) << 16);                           // tensor_dim0 lo16 @ [63:48]
  g1[2] = (int)(((tdim0 >> 16) & 0xFFFFu) | ((tdim1 & 0xFFFFu) << 16)); // dim0 hi / dim1 lo
  g1[3] = (int)(((tdim1 >> 16) & 0xFFFFu) | ((unsigned)BK << 16));  // dim1 hi | tile_dim0=32
  g1[4] = (int)BM;                                                  // tile_dim1=64, tile_dim2=0
  g1[5] = (int)(s0 & 0xFFFFFFFFull);                                // tensor_dim0_stride lo32
  g1[6] = (int)((s0 >> 32) & 0xFFFFull);                            // stride hi16; dim1_stride=0
  g1[7] = 0;
  const int32x4 z4 = {0, 0, 0, 0};
#if __clang_major__ >= 23
  const int32x8 z8 = {0, 0, 0, 0, 0, 0, 0, 0};
  __builtin_amdgcn_tensor_load_to_lds(g0, g1, z4, z4, z8, 0);
#else
  __builtin_amdgcn_tensor_load_to_lds(g0, g1, z4, z4, 0);
#endif
}

// C[M,N] = A[M,K] * B[N,K]^T  (A,B f16 row-major, f32 accumulate) + fused epilogue.
// grid: (ceil(N/64), ceil(M/64), batch); block: 128 threads (4 waves, 2x2).
// Tiles fed by TDM, double-buffered; edges (M,N,K) handled by TDM zero-fill.
__global__ __launch_bounds__(128)
void wmma_gemm(const _Float16* __restrict__ A, int lda, long aStride,
               const _Float16* __restrict__ B, int ldb, long bStride,
               int M, int N, int K, int mode,
               const float* __restrict__ bias,      // per-n
               const float* __restrict__ rowbias,   // per-m (EP_CROSS)
               const float* __restrict__ gamma,     // per-n
               float* outF32, int ldc, long cStride,
               _Float16* outF16)
{
  extern __shared__ _Float16 smem[];   // no static LDS in this kernel -> offset 0

  const int bz = blockIdx.z;
  A += (long)bz * aStride;
  B += (long)bz * bStride;
  if (outF32) outF32 += (long)bz * cStride;
  if (outF16) outF16 += (long)bz * cStride;

  const int blockM = blockIdx.y * BM;
  const int blockN = blockIdx.x * BN;

  const int t       = threadIdx.x;
  const int lane    = t & 31;
  const int wave    = t >> 5;
  const int wm      = wave >> 1;     // 0..1 (M)
  const int wn      = wave & 1;      // 0..1 (N)
  const int halfSel = lane >> 4;     // 0/1
  const int laneLow = lane & 15;
  const bool issuer = (wave == 0);   // wave-uniform branch; TDM is a per-wave op

  const _Float16* Abase = A + (long)blockM * lda;
  const _Float16* Bbase = B + (long)blockN * ldb;
  const int remM = M - blockM;
  const int remN = N - blockN;
  const int nK = (K + BK - 1) / BK;

  f8 acc[2][2];
  const f8 zero = {0.f, 0.f, 0.f, 0.f, 0.f, 0.f, 0.f, 0.f};
#pragma unroll
  for (int i = 0; i < 2; ++i)
#pragma unroll
    for (int j = 0; j < 2; ++j) acc[i][j] = zero;

  if (issuer) {  // prime stage 0
    tdm_load_tile(Abase, lda, remM, K, 0);
    tdm_load_tile(Bbase, ldb, remN, K, TILE_HALVES * 2);
  }

  for (int kt = 0; kt < nK; ++kt) {
    const int buf = kt & 1;
    if (issuer) {
      if (kt + 1 < nK) {  // issue next stage, then retire current (in-order TDM)
        const int k0n = (kt + 1) * BK;
        const unsigned nb = (unsigned)((kt + 1) & 1) * STAGE_BYTES;
        tdm_load_tile(Abase + k0n, lda, remM, K - k0n, nb);
        tdm_load_tile(Bbase + k0n, ldb, remN, K - k0n, nb + TILE_HALVES * 2);
        __builtin_amdgcn_s_wait_tensorcnt(2);
      } else {
        __builtin_amdgcn_s_wait_tensorcnt(0);
      }
    }
    __syncthreads();  // tile `buf` visible to all waves

    const _Float16* As = smem + buf * (2 * TILE_HALVES);
    const _Float16* Bs = As + TILE_HALVES;

    // fragments: ISA 16-bit A 16x32 layout. lane m=laneLow;
    // halfSel=0 -> K{0..7},{16..23}; halfSel=1 -> K{8..15},{24..31}.
    h16 afrag[2], bfrag[2];
#pragma unroll
    for (int wi = 0; wi < 2; ++wi) {
      const _Float16* ap = &As[(wm * 32 + wi * 16 + laneLow) * LDS_STRIDE];
      afrag[wi] = cat16(*(const h8*)(ap + halfSel * 8),
                        *(const h8*)(ap + 16 + halfSel * 8));
      const _Float16* bp = &Bs[(wn * 32 + wi * 16 + laneLow) * LDS_STRIDE];
      bfrag[wi] = cat16(*(const h8*)(bp + halfSel * 8),
                        *(const h8*)(bp + 16 + halfSel * 8));
    }
#pragma unroll
    for (int wi = 0; wi < 2; ++wi)
#pragma unroll
      for (int wj = 0; wj < 2; ++wj)
        acc[wi][wj] = __builtin_amdgcn_wmma_f32_16x16x32_f16(
            false, afrag[wi], false, bfrag[wj], (short)0, acc[wi][wj], false, false);
    __syncthreads();  // safe to let TDM overwrite this buffer next-next stage
  }

  // ---- epilogue: D layout n=laneLow, m=r+8*halfSel ----
#pragma unroll
  for (int wi = 0; wi < 2; ++wi) {
#pragma unroll
    for (int wj = 0; wj < 2; ++wj) {
      const int nIdx = blockN + wn * 32 + wj * 16 + laneLow;
#pragma unroll
      for (int r = 0; r < 8; ++r) {
        const int mIdx = blockM + wm * 32 + wi * 16 + r + halfSel * 8;
        if (mIdx < M && nIdx < N) {
          const float v = acc[wi][wj][r];
          const long o = (long)mIdx * ldc + nIdx;
          switch (mode) {
            case EP_BIAS_F32:
              outF32[o] = v + bias[nIdx];
              break;
            case EP_CROSS:  // x = g1[n]*(acc + ab[m]) + x
              outF32[o] = gamma[nIdx] * (v + rowbias[mIdx]) + outF32[o];
              break;
            case EP_GELU_F16: {  // exact GELU
              const float u = v + bias[nIdx];
              outF16[o] = (_Float16)(0.5f * u * (1.0f + erff(u * 0.70710678118654752f)));
            } break;
            case EP_GAMMA_RES:  // x = g2[n]*(acc + b2[n]) + x
              outF32[o] = gamma[nIdx] * (v + bias[nIdx]) + outF32[o];
              break;
          }
        }
      }
    }
  }
}

// ---------------- fake-quant prep ----------------
__global__ void absmax_scale(const float* __restrict__ w, long sliceElems,
                             float* __restrict__ scales, int scaleOff) {
  __shared__ float red[256];
  const long base = (long)blockIdx.x * sliceElems;
  float m = 0.f;
  for (long i = threadIdx.x; i < sliceElems; i += blockDim.x)
    m = fmaxf(m, fabsf(w[base + i]));
  red[threadIdx.x] = m;
  __syncthreads();
  for (int s = 128; s > 0; s >>= 1) {
    if ((int)threadIdx.x < s) red[threadIdx.x] = fmaxf(red[threadIdx.x], red[threadIdx.x + s]);
    __syncthreads();
  }
  if (threadIdx.x == 0) scales[scaleOff + blockIdx.x] = red[0] / 127.f + 1e-8f;
}

__global__ void quant_f16(const float* __restrict__ w, const float* __restrict__ scales,
                          int scaleOff, long sliceElems, long total,
                          _Float16* __restrict__ out) {
  const long i = (long)blockIdx.x * blockDim.x + threadIdx.x;
  if (i >= total) return;
  const float s = scales[scaleOff + (int)(i / sliceElems)];
  float q = rintf(w[i] / s);                    // round half-to-even, like jnp.round
  q = fminf(fmaxf(q, -128.f), 127.f) * s;
  out[i] = (_Float16)q;
}

// ---------------- data movement / elementwise ----------------
// im2col: out[(b,p)][c*256+iy*16+ix] = x[b][c][py*16+iy][px*16+ix]
__global__ void im2col16(const float* __restrict__ x, _Float16* __restrict__ out) {
  const long i = (long)blockIdx.x * blockDim.x + threadIdx.x;
  const long total = (long)ROWS * CONVK;
  if (i >= total) return;
  const int k = (int)(i % CONVK);
  const long m = i / CONVK;
  const int b = (int)(m / PATCHES), p = (int)(m % PATCHES);
  const int py = p / 14, px = p % 14;
  const int c = k >> 8, iy = (k >> 4) & 15, ix = k & 15;
  const long xi = (((long)b * 3 + c) * 224 + (py * 16 + iy)) * 224 + (px * 16 + ix);
  out[i] = (_Float16)x[xi];
}

// cross-patch B operand: Tt[b][d][p] = x[b][p][d]*a[d]+bsh[d]  (natural K=196)
__global__ void norm1_transpose(const float* __restrict__ x, const float* __restrict__ a,
                                const float* __restrict__ bsh, _Float16* __restrict__ out) {
  const long i = (long)blockIdx.x * blockDim.x + threadIdx.x;
  const long total = (long)BATCH * DIM * PATCHES;
  if (i >= total) return;
  const int p = (int)(i % PATCHES);
  const long r = i / PATCHES;
  const int d = (int)(r % DIM);
  const int b = (int)(r / DIM);
  out[i] = (_Float16)(x[((long)b * PATCHES + p) * DIM + d] * a[d] + bsh[d]);
}

// MLP A operand: out[(b,p)][d] = x*a+b (f16)
__global__ void norm_rows(const float* __restrict__ x, const float* __restrict__ a,
                          const float* __restrict__ b, _Float16* __restrict__ out) {
  const long i = (long)blockIdx.x * blockDim.x + threadIdx.x;
  const long total = (long)ROWS * DIM;
  if (i >= total) return;
  const int d = (int)(i % DIM);
  out[i] = (_Float16)(x[i] * a[d] + b[d]);
}

// final affine norm + mean pool over patches -> pooled f16 [64][384]
__global__ void final_pool(const float* __restrict__ x, const float* __restrict__ na,
                           const float* __restrict__ nb, _Float16* __restrict__ out) {
  const long i = (long)blockIdx.x * blockDim.x + threadIdx.x;
  if (i >= (long)BATCH * DIM) return;
  const int d = (int)(i % DIM);
  const int b = (int)(i / DIM);
  float s = 0.f;
  for (int p = 0; p < PATCHES; ++p)
    s += x[((long)b * PATCHES + p) * DIM + d] * na[d] + nb[d];
  out[i] = (_Float16)(s * (1.0f / PATCHES));
}

// ---------------- host orchestration ----------------
extern "C" void kernel_launch(void* const* d_in, const int* in_sizes, int n_in,
                              void* d_out, int out_size, void* d_ws, size_t ws_size,
                              hipStream_t stream) {
  const float* x       = (const float*)d_in[0];
  const float* conv_w  = (const float*)d_in[1];
  const float* conv_b  = (const float*)d_in[2];
  const float* norm1_a = (const float*)d_in[3];
  const float* norm1_b = (const float*)d_in[4];
  const float* attn_w  = (const float*)d_in[5];
  const float* attn_b  = (const float*)d_in[6];
  const float* gamma1  = (const float*)d_in[7];
  const float* norm2_a = (const float*)d_in[8];
  const float* norm2_b = (const float*)d_in[9];
  const float* mlp_w1  = (const float*)d_in[10];
  const float* mlp_b1  = (const float*)d_in[11];
  const float* mlp_w2  = (const float*)d_in[12];
  const float* mlp_b2  = (const float*)d_in[13];
  const float* gamma2  = (const float*)d_in[14];
  const float* norm_a  = (const float*)d_in[15];
  const float* norm_b  = (const float*)d_in[16];
  const float* head_w  = (const float*)d_in[17];
  const float* head_b  = (const float*)d_in[18];
  float* out = (float*)d_out;

  // workspace layout (256B aligned slots)
  size_t off = 0;
  auto alloc = [&](size_t bytes) { size_t o = off; off = (off + bytes + 255) & ~(size_t)255; return o; };
  char* ws = (char*)d_ws;

  const size_t o_scales = alloc(128 * 4);
  const size_t o_convq  = alloc((size_t)DIM * CONVK * 2);
  const size_t o_headq  = alloc((size_t)NCLS * DIM * 2);
  const size_t o_attnq  = alloc((size_t)NBLK * PATCHES * PATCHES * 2);
  const size_t o_w1q    = alloc((size_t)NBLK * HID * DIM * 2);
  const size_t o_w2q    = alloc((size_t)NBLK * DIM * HID * 2);
  const size_t o_xbuf   = alloc((size_t)ROWS * DIM * 4);
  const size_t o_Tt     = alloc((size_t)BATCH * DIM * PATCHES * 2);
  const size_t o_A2     = alloc((size_t)ROWS * DIM * 2);
  const size_t o_gelu   = alloc((size_t)ROWS * HID * 2);  // also aliased as im2col
  const size_t o_pool   = alloc((size_t)BATCH * DIM * 2);
  if (ws_size < off) return;  // workspace too small; bail deterministically

  float*    scales = (float*)(ws + o_scales);
  _Float16* convq  = (_Float16*)(ws + o_convq);
  _Float16* headq  = (_Float16*)(ws + o_headq);
  _Float16* attnq  = (_Float16*)(ws + o_attnq);
  _Float16* w1q    = (_Float16*)(ws + o_w1q);
  _Float16* w2q    = (_Float16*)(ws + o_w2q);
  float*    xbuf   = (float*)(ws + o_xbuf);
  _Float16* Tt     = (_Float16*)(ws + o_Tt);
  _Float16* A2     = (_Float16*)(ws + o_A2);
  _Float16* gelu   = (_Float16*)(ws + o_gelu);
  _Float16* imcol  = (_Float16*)(ws + o_gelu);  // alias: dead before MLP1 writes
  _Float16* pooled = (_Float16*)(ws + o_pool);

  // ---- quantize weights (per-tensor / per-block-slice symmetric 8-bit) ----
  absmax_scale<<<1, 256, 0, stream>>>(conv_w, (long)DIM * CONVK, scales, 0);
  absmax_scale<<<1, 256, 0, stream>>>(head_w, (long)NCLS * DIM, scales, 1);
  absmax_scale<<<NBLK, 256, 0, stream>>>(attn_w, (long)PATCHES * PATCHES, scales, 2);
  absmax_scale<<<NBLK, 256, 0, stream>>>(mlp_w1, (long)HID * DIM, scales, 26);
  absmax_scale<<<NBLK, 256, 0, stream>>>(mlp_w2, (long)DIM * HID, scales, 50);

  {
    long n = (long)DIM * CONVK;
    quant_f16<<<(unsigned)((n + 255) / 256), 256, 0, stream>>>(conv_w, scales, 0, n, n, convq);
    n = (long)NCLS * DIM;
    quant_f16<<<(unsigned)((n + 255) / 256), 256, 0, stream>>>(head_w, scales, 1, n, n, headq);
    n = (long)NBLK * PATCHES * PATCHES;
    quant_f16<<<(unsigned)((n + 255) / 256), 256, 0, stream>>>(attn_w, scales, 2,
                                                               (long)PATCHES * PATCHES, n, attnq);
    n = (long)NBLK * HID * DIM;
    quant_f16<<<(unsigned)((n + 255) / 256), 256, 0, stream>>>(mlp_w1, scales, 26, (long)HID * DIM, n, w1q);
    quant_f16<<<(unsigned)((n + 255) / 256), 256, 0, stream>>>(mlp_w2, scales, 50, (long)DIM * HID, n, w2q);
  }

  // ---- patch embed: im2col + GEMM -> xbuf[B,196,384] ----
  {
    long n = (long)ROWS * CONVK;
    im2col16<<<(unsigned)((n + 255) / 256), 256, 0, stream>>>(x, imcol);
    dim3 g(DIM / BN, ROWS / BM, 1);  // (6, 196, 1)
    wmma_gemm<<<g, 128, GEMM_LDS_BYTES, stream>>>(
        imcol, CONVK, 0, convq, CONVK, 0,
        ROWS, DIM, CONVK, EP_BIAS_F32,
        conv_b, nullptr, nullptr, xbuf, DIM, 0, nullptr);
  }

  // ---- 24 residual blocks ----
  for (int blk = 0; blk < NBLK; ++blk) {
    // norm1 + transpose -> Tt[b][d][p]
    {
      long n = (long)BATCH * DIM * PATCHES;
      norm1_transpose<<<(unsigned)((n + 255) / 256), 256, 0, stream>>>(
          xbuf, norm1_a + (long)blk * DIM, norm1_b + (long)blk * DIM, Tt);
    }
    // cross-patch: per batch  C[q,d] = W[q,p] * T[p,d]; x += g1*(C+ab[q])
    {
      dim3 g(DIM / BN, (PATCHES + BM - 1) / BM, BATCH);  // (6, 4, 64)
      wmma_gemm<<<g, 128, GEMM_LDS_BYTES, stream>>>(
          attnq + (long)blk * PATCHES * PATCHES, PATCHES, 0,
          Tt, PATCHES, (long)DIM * PATCHES,
          PATCHES, DIM, PATCHES, EP_CROSS,
          nullptr, attn_b + (long)blk * PATCHES, gamma1 + (long)blk * DIM,
          xbuf, DIM, (long)PATCHES * DIM, nullptr);
    }
    // norm2 -> A2[(b,p)][d] f16
    {
      long n = (long)ROWS * DIM;
      norm_rows<<<(unsigned)((n + 255) / 256), 256, 0, stream>>>(
          xbuf, norm2_a + (long)blk * DIM, norm2_b + (long)blk * DIM, A2);
    }
    // MLP1: gelu = GELU(A2 * W1^T + b1)   [12544 x 1536]
    {
      dim3 g(HID / BN, ROWS / BM, 1);  // (24, 196, 1)
      wmma_gemm<<<g, 128, GEMM_LDS_BYTES, stream>>>(
          A2, DIM, 0, w1q + (long)blk * HID * DIM, DIM, 0,
          ROWS, HID, DIM, EP_GELU_F16,
          mlp_b1 + (long)blk * HID, nullptr, nullptr,
          nullptr, HID, 0, gelu);
    }
    // MLP2: x += g2 * (gelu * W2^T + b2)
    {
      dim3 g(DIM / BN, ROWS / BM, 1);  // (6, 196, 1)
      wmma_gemm<<<g, 128, GEMM_LDS_BYTES, stream>>>(
          gelu, HID, 0, w2q + (long)blk * DIM * HID, HID, 0,
          ROWS, DIM, HID, EP_GAMMA_RES,
          mlp_b2 + (long)blk * DIM, nullptr, gamma2 + (long)blk * DIM,
          xbuf, DIM, 0, nullptr);
    }
  }

  // ---- final norm + mean pool + head ----
  {
    long n = (long)BATCH * DIM;
    final_pool<<<(unsigned)((n + 255) / 256), 256, 0, stream>>>(xbuf, norm_a, norm_b, pooled);
    dim3 g((NCLS + BN - 1) / BN, BATCH / BM, 1);  // (16, 1, 1)
    wmma_gemm<<<g, 128, GEMM_LDS_BYTES, stream>>>(
        pooled, DIM, 0, headq, DIM, 0,
        BATCH, NCLS, DIM, EP_BIAS_F32,
        head_b, nullptr, nullptr, out, NCLS, 0, nullptr);
  }
  (void)in_sizes; (void)n_in; (void)out_size;
}
